// TitanTransformer_7610682048662
// MI455X (gfx1250) — matrix operand order repaired and checked by reading, not compiled, not run on hardware
//
#include <hip/hip_runtime.h>
#include <math.h>

// ---------------------------------------------------------------------------
// Types for CDNA5 WMMA (wave32): V_WMMA_F32_16X16X32_BF16
// ---------------------------------------------------------------------------
typedef __bf16 v16bf __attribute__((ext_vector_type(16)));
typedef __bf16 v8bf  __attribute__((ext_vector_type(8)));
typedef __bf16 v4bf  __attribute__((ext_vector_type(4)));
typedef float  v8f   __attribute__((ext_vector_type(8)));
typedef float  v4f   __attribute__((ext_vector_type(4)));

#define EP_GELU 1   // apply exact gelu to (acc+bias)
#define EP_RES  2   // add residual res[m*ldc+n] before store
#define EP_MOE  4   // C += (acc+bias) * rowscale[m*rsStride]  (accumulate)

// ---------------------------------------------------------------------------
// Generic tiled GEMM:  C[M,N] = A[M,K] * op(B) + epilogue
//   transB==0 : op(B)[k,n] = B[n*ldb + k]   (C = A @ B^T, B is [N,K])
//   transB==1 : op(B)[k,n] = B[k*ldb + n]   (C = A @ B,   B is [K,N])
// 256 threads = 8 waves; WG tile 128x64; wave tile 32x32 (2x2 WMMA frags).
// Ping-pong LDS double buffering: one barrier per K-step, next tile staged
// (global_load_b128 -> v_cvt to bf16x4 -> 8B ds_store) before the WMMA burst.
// Requires M%128==0, N%64==0, K%32==0, lda/ldb%4==0 (true for every call).
// ---------------------------------------------------------------------------
#define TM 128
#define TN 64
#define TK 32

__global__ void __launch_bounds__(256)
gemm_wmma(const float* __restrict__ A, int lda,
          const float* __restrict__ B, int ldb, int transB,
          const float* __restrict__ bias,
          const float* res,                   // residual (ld = ldc), may alias C
          const float* __restrict__ rowscale, int rsStride,
          float* C, int ldc,
          int M, int N, int Ksz, int flags)
{
  __shared__ __bf16 As[2][TM][TK];   // 16 KB
  __shared__ __bf16 Bs[2][TN][TK];   //  8 KB

  const int t     = threadIdx.x;
  const int wave  = t >> 5;
  const int lane  = t & 31;
  const int halfl = lane >> 4;     // 0: lanes 0-15, 1: lanes 16-31
  const int l16   = lane & 15;

  const int tileM = blockIdx.y * TM;
  const int tileN = blockIdx.x * TN;
  const int waveM = (wave >> 1) * 32;   // 0,32,64,96
  const int waveN = (wave & 1) * 32;    // 0,32

  // stage one K-tile (A:128x32, B:64x32) into LDS buffer `buf`
  auto stage = [&](int buf, int k0) {
    // A: 1024 float4 groups, 4 per thread: global_load_b128 -> bf16x4 -> b64 store
#pragma unroll
    for (int j = 0; j < 4; ++j) {
      int idx4 = t + 256 * j;
      int r = idx4 >> 3;            // row 0..127
      int c = (idx4 & 7) * 4;       // col 0,4,..,28
      const v4f v = *(const v4f*)&A[(size_t)(tileM + r) * lda + (k0 + c)];
      *(v4bf*)&As[buf][r][c] = __builtin_convertvector(v, v4bf);
    }
    if (transB) {
      // scalar, coalesced over n (only used by the small P*V GEMMs)
#pragma unroll
      for (int j = 0; j < 8; ++j) {
        int idx = t + 256 * j;
        int n = idx & 63, c = idx >> 6;
        Bs[buf][n][c] = (__bf16)B[(size_t)(k0 + c) * ldb + (tileN + n)];
      }
    } else {
#pragma unroll
      for (int j = 0; j < 2; ++j) {
        int idx4 = t + 256 * j;
        int r = idx4 >> 3;          // row 0..63
        int c = (idx4 & 7) * 4;
        const v4f v = *(const v4f*)&B[(size_t)(tileN + r) * ldb + (k0 + c)];
        *(v4bf*)&Bs[buf][r][c] = __builtin_convertvector(v, v4bf);
      }
    }
  };

  v8f acc[2][2] = {};
  const int nk = Ksz / TK;

  stage(0, 0);
  __syncthreads();

  for (int kt = 0; kt < nk; ++kt) {
    const int cur = kt & 1;
    // stage next tile into the other buffer (overlaps with WMMA below)
    if (kt + 1 < nk) stage(cur ^ 1, (kt + 1) * TK);
    if (kt + 2 < nk)
      __builtin_prefetch(&A[(size_t)(tileM + (t & 127)) * lda + (kt + 2) * TK], 0, 3);

    // ---- fragments per ISA VGPR layouts ----
    // A 16x32 bf16: lane l: m=l%16; e<8 -> k=half*8+e ; e>=8 -> k=16+half*8+(e-8)
    // B 32x16 bf16: lane l: n=l%16; e<8 -> k=half*8+e ; e>=8 -> k=16+half*8+(e-8)
    v16bf af[2], bfr[2];
#pragma unroll
    for (int i = 0; i < 2; ++i) {
      int m = waveM + i * 16 + l16;
      v8bf alo = *(const v8bf*)&As[cur][m][halfl * 8];
      v8bf ahi = *(const v8bf*)&As[cur][m][16 + halfl * 8];
      af[i] = __builtin_shufflevector(alo, ahi, 0,1,2,3,4,5,6,7,8,9,10,11,12,13,14,15);
      int n = waveN + i * 16 + l16;
      v8bf blo = *(const v8bf*)&Bs[cur][n][halfl * 8];
      v8bf bhi = *(const v8bf*)&Bs[cur][n][16 + halfl * 8];
      bfr[i] = __builtin_shufflevector(blo, bhi, 0,1,2,3,4,5,6,7,8,9,10,11,12,13,14,15);
    }
#pragma unroll
    for (int i = 0; i < 2; ++i)
#pragma unroll
      for (int j = 0; j < 2; ++j)
        acc[i][j] = __builtin_amdgcn_wmma_f32_16x16x32_bf16(
            false, af[i], false, bfr[j], (short)0, acc[i][j], false, false);

    __syncthreads();   // single barrier per K-step (ping-pong)
  }

  // ---- epilogue; C layout: elem e, lane l -> m = e + 8*(l/16), n = l%16 ----
#pragma unroll
  for (int i = 0; i < 2; ++i)
#pragma unroll
    for (int j = 0; j < 2; ++j)
#pragma unroll
      for (int e = 0; e < 8; ++e) {
        int m = tileM + waveM + i * 16 + e + 8 * halfl;
        int n = tileN + waveN + j * 16 + l16;
        float v = acc[i][j][e];
        if (bias) v += bias[n];
        if (flags & EP_GELU) v = 0.5f * v * (1.0f + erff(v * 0.70710678118654752f));
        size_t off = (size_t)m * ldc + n;
        if (flags & EP_MOE) {
          C[off] += v * rowscale[(size_t)m * rsStride];
        } else {
          if (flags & EP_RES) v += res[off];
          C[off] = v;
        }
      }
}

// ---------------------------------------------------------------------------
// Elementwise / reduction kernels
// ---------------------------------------------------------------------------
__global__ void __launch_bounds__(256)
embed_kernel(const int* __restrict__ ids, const float* __restrict__ ew,
             const float* __restrict__ pw, float* __restrict__ x, int total)
{
  int idx = blockIdx.x * 256 + threadIdx.x;
  if (idx >= total) return;
  int tkn = idx >> 10, d = idx & 1023;               // D = 1024
  x[idx] = ew[(size_t)ids[tkn] * 1024 + d] + pw[(size_t)tkn * 1024 + d];
}

__global__ void __launch_bounds__(256)
ln_kernel(const float* __restrict__ x, const float* __restrict__ g,
          const float* __restrict__ b, float* __restrict__ y, int n)
{
  __shared__ float red[256];
  const int row = blockIdx.x, t = threadIdx.x;
  const float* xr = x + (size_t)row * n;
  float s = 0.f;
  for (int d = t; d < n; d += 256) s += xr[d];
  red[t] = s; __syncthreads();
  for (int o = 128; o > 0; o >>= 1) { if (t < o) red[t] += red[t + o]; __syncthreads(); }
  const float mean = red[0] / n;
  __syncthreads();
  float v = 0.f;
  for (int d = t; d < n; d += 256) { float dd = xr[d] - mean; v += dd * dd; }
  red[t] = v; __syncthreads();
  for (int o = 128; o > 0; o >>= 1) { if (t < o) red[t] += red[t + o]; __syncthreads(); }
  const float rstd = rsqrtf(red[0] / n + 1e-5f);
  for (int d = t; d < n; d += 256)
    y[(size_t)row * n + d] = (xr[d] - mean) * rstd * g[d] + b[d];
}

__global__ void __launch_bounds__(256)
softmax_kernel(float* __restrict__ s, int n, float scale)
{
  __shared__ float red[256];
  const int row = blockIdx.x, t = threadIdx.x;
  float* p = s + (size_t)row * n;
  float mx = -3.0e38f;
  for (int i = t; i < n; i += 256) mx = fmaxf(mx, p[i] * scale);
  red[t] = mx; __syncthreads();
  for (int o = 128; o > 0; o >>= 1) { if (t < o) red[t] = fmaxf(red[t], red[t + o]); __syncthreads(); }
  mx = red[0];
  __syncthreads();
  float sum = 0.f;
  for (int i = t; i < n; i += 256) { float e = expf(p[i] * scale - mx); p[i] = e; sum += e; }
  red[t] = sum; __syncthreads();
  for (int o = 128; o > 0; o >>= 1) { if (t < o) red[t] += red[t + o]; __syncthreads(); }
  const float inv = 1.f / red[0];
  for (int i = t; i < n; i += 256) p[i] *= inv;
}

// gate scores: gs[token,e] = h[token,:] . gate_w[e,:] + gb[e]   (D=1024, E=8)
__global__ void __launch_bounds__(64)
gate_kernel(const float* __restrict__ h, const float* __restrict__ gw,
            const float* __restrict__ gb, float* __restrict__ gs)
{
  __shared__ float red[64];
  const int tkn = blockIdx.x;
  const int e = threadIdx.x >> 3, p = threadIdx.x & 7;
  float s = 0.f;
  const float* hr = h + (size_t)tkn * 1024;
  const float* wr = gw + (size_t)e * 1024;
  for (int k = p * 128; k < p * 128 + 128; ++k) s += hr[k] * wr[k];
  red[threadIdx.x] = s; __syncthreads();
  if (p == 0) {
    float tot = 0.f;
    for (int q = 0; q < 8; ++q) tot += red[e * 8 + q];
    gs[(size_t)tkn * 8 + e] = tot + gb[e];
  }
}

// top-2 over E=8 + softmax over the two -> dense combine [S,E]
__global__ void __launch_bounds__(256)
topk_kernel(const float* __restrict__ gs, float* __restrict__ comb, int S)
{
  int tkn = blockIdx.x * 256 + threadIdx.x;
  if (tkn >= S) return;
  float s[8];
  for (int e = 0; e < 8; ++e) s[e] = gs[(size_t)tkn * 8 + e];
  int i0 = 0;
  for (int e = 1; e < 8; ++e) if (s[e] > s[i0]) i0 = e;
  int i1 = -1;
  for (int e = 0; e < 8; ++e) { if (e == i0) continue; if (i1 < 0 || s[e] > s[i1]) i1 = e; }
  float m  = s[i0];
  float e0 = 1.f;                       // exp(s[i0]-m)
  float e1 = expf(s[i1] - m);
  float inv = 1.f / (e0 + e1);
  for (int e = 0; e < 8; ++e)
    comb[(size_t)tkn * 8 + e] = (e == i0) ? e0 * inv : (e == i1) ? e1 * inv : 0.f;
}

// ---------------------------------------------------------------------------
// Host orchestration
// ---------------------------------------------------------------------------
static inline void launch_gemm(hipStream_t st,
    const float* A, int lda, const float* B, int ldb, int trB,
    const float* bias, const float* res,
    const float* rscale, int rstride,
    float* C, int ldc, int M, int N, int K, int flags)
{
  dim3 grid(N / TN, M / TM);
  gemm_wmma<<<grid, 256, 0, st>>>(A, lda, B, ldb, trB, bias, res,
                                  rscale, rstride, C, ldc, M, N, K, flags);
}

extern "C" void kernel_launch(void* const* d_in, const int* in_sizes, int n_in,
                              void* d_out, int out_size, void* d_ws, size_t ws_size,
                              hipStream_t stream) {
  (void)in_sizes; (void)n_in; (void)out_size; (void)ws_size;
  const int S = 2048, D = 1024, H = 16, HD = 64, FF = 4096, E = 8, VO = 32000;

  const int*   input_ids = (const int*)  d_in[0];
  const float* embed_w   = (const float*)d_in[1];
  const float* pos_w     = (const float*)d_in[2];
  struct Blk { const float *ln1_g, *ln1_b, *in_w, *in_b, *out_w, *out_b, *ln2_g, *ln2_b; } blk[2];
  int p = 3;
  for (int bi = 0; bi < 2; ++bi) {
    blk[bi].ln1_g = (const float*)d_in[p++]; blk[bi].ln1_b = (const float*)d_in[p++];
    blk[bi].in_w  = (const float*)d_in[p++]; blk[bi].in_b  = (const float*)d_in[p++];
    blk[bi].out_w = (const float*)d_in[p++]; blk[bi].out_b = (const float*)d_in[p++];
    blk[bi].ln2_g = (const float*)d_in[p++]; blk[bi].ln2_b = (const float*)d_in[p++];
  }
  const float* ffn_w1 = (const float*)d_in[19];
  const float* ffn_b1 = (const float*)d_in[20];
  const float* ffn_w2 = (const float*)d_in[21];
  const float* ffn_b2 = (const float*)d_in[22];
  const float* gate_w = (const float*)d_in[23];
  const float* gate_b = (const float*)d_in[24];
  const float* moe_w1 = (const float*)d_in[25];
  const float* moe_b1 = (const float*)d_in[26];
  const float* moe_w2 = (const float*)d_in[27];
  const float* moe_b2 = (const float*)d_in[28];
  const float* norm_g = (const float*)d_in[29];
  const float* norm_b = (const float*)d_in[30];
  const float* head_w = (const float*)d_in[31];

  // workspace carve-out (floats): ~96 MB total
  float* ws      = (float*)d_ws;
  float* x       = ws;                                  // [S,D]
  float* h       = x       + (size_t)S * D;             // [S,D]
  float* qkv     = h       + (size_t)S * D;             // [S,3D]
  float* attnout = qkv     + (size_t)S * 3 * D;         // [S,D]
  float* scores  = attnout + (size_t)S * D;             // [S,S] (per-head reuse)
  float* hid     = scores  + (size_t)S * S;             // [S,FF]
  float* gs      = hid     + (size_t)S * FF;            // [S,E]
  float* comb    = gs      + (size_t)S * E;             // [S,E]

  // embedding + positional
  embed_kernel<<<(S * D + 255) / 256, 256, 0, stream>>>(input_ids, embed_w, pos_w, x, S * D);

  for (int bi = 0; bi < 2; ++bi) {
    const Blk& L = blk[bi];
    // ---- MHA ----
    ln_kernel<<<S, 256, 0, stream>>>(x, L.ln1_g, L.ln1_b, h, D);
    launch_gemm(stream, h, D, L.in_w, D, 0, L.in_b, nullptr, nullptr, 0,
                qkv, 3 * D, S, 3 * D, D, 0);
    for (int hh = 0; hh < H; ++hh) {
      const float* Q  = qkv + hh * HD;
      const float* Kp = qkv + D + hh * HD;
      const float* Vp = qkv + 2 * D + hh * HD;
      launch_gemm(stream, Q, 3 * D, Kp, 3 * D, 0, nullptr, nullptr, nullptr, 0,
                  scores, S, S, S, HD, 0);                      // S = Q K^T
      softmax_kernel<<<S, 256, 0, stream>>>(scores, S, 0.125f); // 1/sqrt(64)
      launch_gemm(stream, scores, S, Vp, 3 * D, 1, nullptr, nullptr, nullptr, 0,
                  attnout + hh * HD, D, S, HD, S, 0);           // O = P V
    }
    launch_gemm(stream, attnout, D, L.out_w, D, 0, L.out_b, x, nullptr, 0,
                x, D, S, D, D, EP_RES);                         // x += proj

    // ---- FFN / MoE ----
    ln_kernel<<<S, 256, 0, stream>>>(x, L.ln2_g, L.ln2_b, h, D);
    if (bi == 0) {
      launch_gemm(stream, h, D, ffn_w1, D, 0, ffn_b1, nullptr, nullptr, 0,
                  hid, FF, S, FF, D, EP_GELU);
      launch_gemm(stream, hid, FF, ffn_w2, FF, 0, ffn_b2, x, nullptr, 0,
                  x, D, S, D, FF, EP_RES);                      // x += ffn
    } else {
      gate_kernel<<<S, 64, 0, stream>>>(h, gate_w, gate_b, gs);
      topk_kernel<<<(S + 255) / 256, 256, 0, stream>>>(gs, comb, S);
      for (int e = 0; e < E; ++e) {
        launch_gemm(stream, h, D, moe_w1 + (size_t)e * FF * D, D, 0,
                    moe_b1 + (size_t)e * FF, nullptr, nullptr, 0,
                    hid, FF, S, FF, D, EP_GELU);
        launch_gemm(stream, hid, FF, moe_w2 + (size_t)e * D * FF, FF, 0,
                    moe_b2 + (size_t)e * D, nullptr, comb + e, E,
                    x, D, S, D, FF, EP_MOE);                    // x += scale*(.)
      }
    }
  }

  // final norm + LM head
  ln_kernel<<<S, 256, 0, stream>>>(x, norm_g, norm_b, h, D);
  launch_gemm(stream, h, D, head_w, D, 0, nullptr, nullptr, nullptr, 0,
              (float*)d_out, VO, S, VO, D, 0);
}
